// TransformerBlock_68410239091300
// MI455X (gfx1250) — matrix operand order repaired and checked by reading, not compiled, not run on hardware
//
#include <hip/hip_runtime.h>
#include <math.h>
#include <stdint.h>

// ---------------- problem dims ----------------
#define D_MODEL 1024
#define N_HEADS 16
#define HEAD_DIM 64
#define D_FF 4096
#define SEQ 2048
#define BATCH 4
#define MROWS (BATCH * SEQ)   // 8192

// ---------------- CDNA5 WMMA types ----------------
typedef __bf16 bf16;
typedef __attribute__((ext_vector_type(16))) __bf16 bf16x16;
typedef __attribute__((ext_vector_type(8)))  float  f32x8;
typedef int v4i __attribute__((vector_size(16)));   // matches builtin's int4 pointee

// ---------------- async global->LDS (CDNA5 GLOBAL_LOAD_ASYNC_TO_LDS_B128) ----------------
#if defined(__has_builtin)
#if __has_builtin(__builtin_amdgcn_global_load_async_to_lds_b128) && \
    __has_builtin(__builtin_amdgcn_s_wait_asynccnt)
#define USE_ASYNC_LDS 1
#endif
#endif

__device__ __forceinline__ void lds_copy16(void* lds, const void* glb) {
#ifdef USE_ASYNC_LDS
  // per-lane 16B async DMA into LDS, tracked by ASYNCcnt
  __builtin_amdgcn_global_load_async_to_lds_b128(
      (__attribute__((address_space(1))) v4i*)(uintptr_t)glb,
      (__attribute__((address_space(3))) v4i*)(uint32_t)(uintptr_t)lds,
      0, 0);
#else
  *reinterpret_cast<uint4*>(lds) = *reinterpret_cast<const uint4*>(glb);
#endif
}
__device__ __forceinline__ void lds_copy_fence() {
#ifdef USE_ASYNC_LDS
  __builtin_amdgcn_s_wait_asynccnt(0);
#endif
}

__device__ __forceinline__ f32x8 wmma_bf16(bf16x16 a, bf16x16 b, f32x8 c) {
  // D = A(16x32) * B(32x16) + C, f32 accumulate
  return __builtin_amdgcn_wmma_f32_16x16x32_bf16(
      /*neg_a=*/false, a, /*neg_b=*/false, b,
      /*c_mod=*/(short)0, c, /*reuse_a=*/false, /*reuse_b=*/false);
}

// ---- fragment loaders (layouts per CDNA5 ISA §7.12.2, wave32) ----
// A 16x32 bf16: lane&15 = row M; lanes 0-15: v0-3 hold K=0..7, v4-7 hold K=16..23;
// lanes 16-31: K ranges shifted by +8.  All reads contiguous -> 2x ds_load_b128.
__device__ __forceinline__ void load_a_frag(bf16x16& a, const bf16* __restrict__ src,
                                            int stride, int lane) {
  const int half = (lane >> 4) & 1;
  const bf16* p = src + (size_t)(lane & 15) * stride;
#pragma unroll
  for (int v = 0; v < 8; ++v) {
    const int k0 = ((v < 4) ? (v * 2) : (16 + (v - 4) * 2)) + half * 8;
    a[2 * v]     = p[k0];
    a[2 * v + 1] = p[k0 + 1];
  }
}

// B fragment from a transposed source laid out [N][K] (row n holds K contiguous):
// lane&15 = B column n; lanes 0-15 hold K=0..7,16..23, lanes 16-31 hold K=8..15,24..31.
// Per lane: 2 runs of 8 contiguous bf16 -> 2x ds_load_b128.
__device__ __forceinline__ void load_bT_frag(bf16x16& b, const bf16* __restrict__ src,
                                             int stride, int lane) {
  const int half = (lane >> 4) & 1;
  const bf16* p = src + (size_t)(lane & 15) * stride;
#pragma unroll
  for (int v = 0; v < 8; ++v) {
    const int k = ((v < 4) ? (v * 2) : (16 + (v - 4) * 2)) + half * 8;
    b[2 * v]     = p[k];
    b[2 * v + 1] = p[k + 1];
  }
}

// ---------------- 16-lane (half-wave) reductions for softmax rows ----------------
__device__ __forceinline__ float redmax16(float x) {
#pragma unroll
  for (int off = 8; off; off >>= 1) x = fmaxf(x, __shfl_xor(x, off, 16));
  return x;
}
__device__ __forceinline__ float redsum16(float x) {
#pragma unroll
  for (int off = 8; off; off >>= 1) x += __shfl_xor(x, off, 16);
  return x;
}

// ------- fp32 [K][N] -> bf16 [N][K] transposed weight conversion (tiled) -------
__global__ __launch_bounds__(256) void cvt_transpose_bf16(const float* __restrict__ src,
                                                          bf16* __restrict__ dst,
                                                          int K, int N) {
  __shared__ __align__(16) bf16 t[64][72];
  const int k0 = blockIdx.y * 64, n0 = blockIdx.x * 64;
  const int tid = threadIdx.x;
#pragma unroll
  for (int i = 0; i < 4; ++i) {
    const int u = tid + i * 256;              // 0..1023, 4 floats each
    const int r = u >> 4, c4 = (u & 15) * 4;  // r = k row, c4 = n col quad
    const float4 f = *reinterpret_cast<const float4*>(src + (size_t)(k0 + r) * N + n0 + c4);
    t[c4 + 0][r] = (bf16)f.x;
    t[c4 + 1][r] = (bf16)f.y;
    t[c4 + 2][r] = (bf16)f.z;
    t[c4 + 3][r] = (bf16)f.w;
  }
  __syncthreads();
#pragma unroll
  for (int i = 0; i < 2; ++i) {
    const int u = tid * 2 + i;                // 0..511 uint4
    const int r = u >> 3, c = (u & 7) * 8;    // r = n row, c = k col
    *reinterpret_cast<uint4*>(dst + (size_t)(n0 + r) * K + k0 + c) =
        *reinterpret_cast<const uint4*>(&t[r][c]);
  }
}

// ------- V [B*S, D] bf16 -> V^T [B,H,HEAD_DIM,S] bf16 (per-head tiled transpose) -------
__global__ __launch_bounds__(128) void transpose_v(const bf16* __restrict__ vsrc,
                                                   bf16* __restrict__ vt) {
  __shared__ __align__(16) bf16 t[64][72];
  const int s0 = blockIdx.x * 64;
  const int bh = blockIdx.y;                 // b*H + h
  const int b = bh >> 4, h = bh & 15;
  const int tid = threadIdx.x;
  const bf16* src = vsrc + ((size_t)b * SEQ) * D_MODEL + h * HEAD_DIM;
#pragma unroll
  for (int i = 0; i < 4; ++i) {
    const int u = tid + i * 128;             // 0..511 uint4 (8 bf16)
    const int r = u >> 3, c = (u & 7) * 8;   // r = s row, c = dh col
    const uint4 d = *reinterpret_cast<const uint4*>(src + (size_t)(s0 + r) * D_MODEL + c);
    const bf16* e = reinterpret_cast<const bf16*>(&d);
#pragma unroll
    for (int q = 0; q < 8; ++q) t[c + q][r] = e[q];
  }
  __syncthreads();
  bf16* dst = vt + ((size_t)bh * HEAD_DIM) * SEQ + s0;
#pragma unroll
  for (int i = 0; i < 4; ++i) {
    const int u = tid + i * 128;             // 0..511 uint4
    const int r = u >> 3, c = (u & 7) * 8;   // r = dh row, c = s col
    *reinterpret_cast<uint4*>(dst + (size_t)r * SEQ + c) =
        *reinterpret_cast<const uint4*>(&t[r][c]);
  }
}

// ---------------- LayerNorm (block per row of 1024) fused with bf16 cast ----------------
__global__ __launch_bounds__(256) void ln_bf16(const float* __restrict__ x,
                                               const float* __restrict__ g,
                                               const float* __restrict__ beta,
                                               bf16* __restrict__ out) {
  const int row = blockIdx.x;
  const int tid = threadIdx.x;
  const float* xr = x + (size_t)row * D_MODEL;
  float vals[4];
  float s = 0.f, s2 = 0.f;
#pragma unroll
  for (int i = 0; i < 4; ++i) {
    const float t = xr[tid + i * 256];
    vals[i] = t; s += t; s2 += t * t;
  }
#pragma unroll
  for (int off = 16; off; off >>= 1) {
    s  += __shfl_xor(s, off, 32);
    s2 += __shfl_xor(s2, off, 32);
  }
  __shared__ float rs[8], rs2[8], bc[2];
  const int wid = tid >> 5, lane = tid & 31;
  if (lane == 0) { rs[wid] = s; rs2[wid] = s2; }
  __syncthreads();
  if (tid == 0) {
    float S = 0.f, S2 = 0.f;
#pragma unroll
    for (int i = 0; i < 8; ++i) { S += rs[i]; S2 += rs2[i]; }
    const float mu  = S * (1.0f / D_MODEL);
    const float var = S2 * (1.0f / D_MODEL) - mu * mu;
    bc[0] = mu; bc[1] = rsqrtf(var + 1e-5f);
  }
  __syncthreads();
  const float mu = bc[0], rsig = bc[1];
  bf16* orow = out + (size_t)row * D_MODEL;
#pragma unroll
  for (int i = 0; i < 4; ++i) {
    const int c = tid + i * 256;
    orow[c] = (bf16)((vals[i] - mu) * rsig * g[c] + beta[c]);
  }
}

// ---------------- generic bf16 WMMA GEMM (weights pre-transposed) ----------------
// C[M,N] = A[M,K](bf16) * Wt[N,K](bf16)^T + bias, with epilogues:
//   EPI=0: bf16 out            EPI=1: exact GELU -> bf16 out
//   EPI=2: + residual -> f32 out
// Block: 256 thr (8 waves), tile 128x128, K-step 32; waves 2(M)x4(N), wave tile 64x32.
template <int EPI>
__global__ __launch_bounds__(256) void gemm_bf16(
    const bf16* __restrict__ A, const bf16* __restrict__ Wt,
    const float* __restrict__ bias, const float* __restrict__ resid,
    bf16* __restrict__ outH, float* __restrict__ outF,
    int M, int N, int K) {
  __shared__ __align__(16) bf16 As[128][40];   // [m][k] 32+8 pad (row 80B, 16B aligned)
  __shared__ __align__(16) bf16 Bts[128][40];  // [n][k] transposed weights
  const int tid = threadIdx.x, lane = tid & 31, w = tid >> 5;
  const int wm = w >> 2, wn = w & 3;
  const int m0 = blockIdx.y * 128, n0 = blockIdx.x * 128;
  f32x8 acc[4][2] = {};

  for (int k0 = 0; k0 < K; k0 += 32) {
#ifdef USE_ASYNC_LDS
#pragma unroll
    for (int i = 0; i < 2; ++i) {
      const int u = tid * 2 + i;            // 0..511
      const int r = u >> 2, q = u & 3;      // 4 uint4 per 32-wide K row
      lds_copy16(&As[r][q * 8],  A  + (size_t)(m0 + r) * K + k0 + q * 8);
      lds_copy16(&Bts[r][q * 8], Wt + (size_t)(n0 + r) * K + k0 + q * 8);
    }
    lds_copy_fence();
#else
    uint4 ta[2], tb[2];
#pragma unroll
    for (int i = 0; i < 2; ++i) {           // batch loads: keep 4 b128 in flight
      const int u = tid * 2 + i;
      const int r = u >> 2, q = u & 3;
      ta[i] = *(reinterpret_cast<const uint4*>(A  + (size_t)(m0 + r) * K + k0) + q);
      tb[i] = *(reinterpret_cast<const uint4*>(Wt + (size_t)(n0 + r) * K + k0) + q);
    }
#pragma unroll
    for (int i = 0; i < 2; ++i) {
      const int u = tid * 2 + i;
      const int r = u >> 2, q = u & 3;
      *reinterpret_cast<uint4*>(&As[r][q * 8])  = ta[i];
      *reinterpret_cast<uint4*>(&Bts[r][q * 8]) = tb[i];
    }
#endif
    if (k0 + 32 < K) {   // hint next K-tile (global_prefetch_b8)
      __builtin_prefetch(A  + (size_t)(m0 + (tid >> 1)) * K + k0 + 32, 0, 0);
      __builtin_prefetch(Wt + (size_t)(n0 + (tid >> 1)) * K + k0 + 32, 0, 0);
    }
    __syncthreads();

    bf16x16 bfr[2];
#pragma unroll
    for (int tj = 0; tj < 2; ++tj)
      load_bT_frag(bfr[tj], &Bts[wn * 32 + tj * 16][0], 40, lane);
#pragma unroll
    for (int ti = 0; ti < 4; ++ti) {
      bf16x16 afr;
      load_a_frag(afr, &As[wm * 64 + ti * 16][0], 40, lane);
#pragma unroll
      for (int tj = 0; tj < 2; ++tj)
        acc[ti][tj] = wmma_bf16(afr, bfr[tj], acc[ti][tj]);
    }
    __syncthreads();
  }

  // epilogue: C/D layout -> row = v + 8*half, col = lane&15
  const int half = lane >> 4, cn = lane & 15;
#pragma unroll
  for (int ti = 0; ti < 4; ++ti) {
#pragma unroll
    for (int tj = 0; tj < 2; ++tj) {
      const int gn = n0 + wn * 32 + tj * 16 + cn;
      const float bv = bias[gn];
#pragma unroll
      for (int v = 0; v < 8; ++v) {
        const int gm = m0 + wm * 64 + ti * 16 + half * 8 + v;
        float val = acc[ti][tj][v] + bv;
        const size_t idx = (size_t)gm * N + gn;
        if (EPI == 0) {
          outH[idx] = (bf16)val;
        } else if (EPI == 1) {
          val = 0.5f * val * (1.0f + erff(val * 0.70710678118654752f));  // exact GELU
          outH[idx] = (bf16)val;
        } else {
          outF[idx] = val + resid[idx];
        }
      }
    }
  }
}

// ---------------- causal flash attention (bf16 WMMA, f32 accum) ----------------
// grid (SEQ/64, BATCH*N_HEADS), 128 thr = 4 waves; wave owns 16 query rows.
// Q/K layout: [B*S, D_MODEL], head h = cols [h*64, h*64+64).  V^T: [b,h,dh,s].
__global__ __launch_bounds__(128) void attn_kernel(const bf16* __restrict__ Q,
                                                   const bf16* __restrict__ Kb,
                                                   const bf16* __restrict__ Vt,
                                                   bf16* __restrict__ O) {
  __shared__ __align__(16) bf16 Ks[64][72];     // [key][dh]
  __shared__ __align__(16) bf16 VsT[64][72];    // [dh][key]
  __shared__ __align__(16) bf16 Pb[4][16][64];  // per-wave probs staging
  const int tid = threadIdx.x, lane = tid & 31, wave = tid >> 5;
  const int bh = blockIdx.y;
  const int b = bh >> 4, h = bh & 15;
  const int qblk = blockIdx.x;
  const int q0 = qblk * 64 + wave * 16;
  const size_t base = ((size_t)b * SEQ) * D_MODEL + (size_t)h * HEAD_DIM;
  const bf16* Qp = Q + base;
  const bf16* Kp = Kb + base;
  const bf16* Vp = Vt + (size_t)bh * HEAD_DIM * SEQ;
  const int half = lane >> 4, cn = lane & 15;

  // Q fragments reused across the whole key loop: load once from global.
  bf16x16 qa[2];
#pragma unroll
  for (int kk = 0; kk < 2; ++kk)
    load_a_frag(qa[kk], Qp + (size_t)q0 * D_MODEL + kk * 32, D_MODEL, lane);

  float m_i[8], l_i[8];
#pragma unroll
  for (int v = 0; v < 8; ++v) { m_i[v] = -1.0e30f; l_i[v] = 0.0f; }
  f32x8 oacc[4] = {};

  for (int j = 0; j <= qblk; ++j) {            // uniform across all waves in block
    __syncthreads();                           // previous iter done with Ks/VsT
#pragma unroll
    for (int i = 0; i < 4; ++i) {              // stage K [key][dh], V^T [dh][key]
      const int u = tid + i * 128;             // 0..511 uint4
      const int r = u >> 3, c = u & 7;
      lds_copy16(&Ks[r][c * 8],  Kp + (size_t)(j * 64 + r) * D_MODEL + c * 8);
      lds_copy16(&VsT[r][c * 8], Vp + (size_t)r * SEQ + j * 64 + c * 8);
    }
    lds_copy_fence();
    __syncthreads();

    // S = Q * K^T : 16 q-rows x 64 keys, Dh=64 -> 2 K-steps x 4 key subtiles
    f32x8 s[4] = {};
#pragma unroll
    for (int kk = 0; kk < 2; ++kk) {
#pragma unroll
      for (int tj = 0; tj < 4; ++tj) {
        bf16x16 kb;
        load_bT_frag(kb, &Ks[tj * 16][kk * 32], 72, lane);
        s[tj] = wmma_bf16(qa[kk], kb, s[tj]);
      }
    }

    // online softmax (finite -1e30 mask keeps control flow uniform; fully-masked
    // tiles contribute exactly 0 because m_i is finite after tile j=0)
#pragma unroll
    for (int v = 0; v < 8; ++v) {
      const int qrow = q0 + half * 8 + v;
      float rmax = -1.0e30f;
#pragma unroll
      for (int tj = 0; tj < 4; ++tj) {
        const int key = j * 64 + tj * 16 + cn;
        float sv = s[tj][v] * 0.125f;          // 1/sqrt(64)
        sv = (key > qrow) ? -1.0e30f : sv;     // causal mask
        s[tj][v] = sv;
        rmax = fmaxf(rmax, sv);
      }
      rmax = redmax16(rmax);
      const float mnew = fmaxf(m_i[v], rmax);
      const float alpha = __expf(m_i[v] - mnew);
      m_i[v] = mnew;
      float rsum = 0.f;
#pragma unroll
      for (int tj = 0; tj < 4; ++tj) {
        const float p = __expf(s[tj][v] - mnew);
        s[tj][v] = p;
        rsum += p;
      }
      l_i[v] = l_i[v] * alpha + redsum16(rsum);
#pragma unroll
      for (int tj = 0; tj < 4; ++tj) oacc[tj][v] *= alpha;
    }

    // stage P to LDS to reshape C-layout -> A-fragment layout
#pragma unroll
    for (int tj = 0; tj < 4; ++tj)
#pragma unroll
      for (int v = 0; v < 8; ++v)
        Pb[wave][half * 8 + v][tj * 16 + cn] = (bf16)s[tj][v];
    __syncthreads();

    // O += P(16x64) * V(64x64); B operand from V^T rows -> contiguous b128 reads
#pragma unroll
    for (int kk = 0; kk < 2; ++kk) {
      bf16x16 pa;
      load_a_frag(pa, &Pb[wave][0][kk * 32], 64, lane);
#pragma unroll
      for (int tj = 0; tj < 4; ++tj) {
        bf16x16 vb;
        load_bT_frag(vb, &VsT[tj * 16][kk * 32], 72, lane);
        oacc[tj] = wmma_bf16(pa, vb, oacc[tj]);
      }
    }
  }

  // normalize and write [B*S, D] bf16
#pragma unroll
  for (int tj = 0; tj < 4; ++tj)
#pragma unroll
    for (int v = 0; v < 8; ++v) {
      const int qrow = q0 + half * 8 + v;
      const float o = oacc[tj][v] / l_i[v];
      O[((size_t)b * SEQ + qrow) * D_MODEL + h * HEAD_DIM + tj * 16 + cn] = (bf16)o;
    }
}

// ---------------- host-side launcher ----------------
extern "C" void kernel_launch(void* const* d_in, const int* in_sizes, int n_in,
                              void* d_out, int out_size, void* d_ws, size_t ws_size,
                              hipStream_t stream) {
  (void)in_sizes; (void)n_in; (void)out_size; (void)ws_size;
  const float* x   = (const float*)d_in[0];
  const float* wq  = (const float*)d_in[1];
  const float* bq  = (const float*)d_in[2];
  const float* wk  = (const float*)d_in[3];
  const float* bk  = (const float*)d_in[4];
  const float* wv  = (const float*)d_in[5];
  const float* bv  = (const float*)d_in[6];
  const float* wo  = (const float*)d_in[7];
  const float* bo  = (const float*)d_in[8];
  const float* w1  = (const float*)d_in[9];
  const float* b1  = (const float*)d_in[10];
  const float* w2  = (const float*)d_in[11];
  const float* b2  = (const float*)d_in[12];
  const float* g1  = (const float*)d_in[13];
  const float* be1 = (const float*)d_in[14];
  const float* g2  = (const float*)d_in[15];
  const float* be2 = (const float*)d_in[16];
  float* out = (float*)d_out;

  // workspace layout (~235 MB), 256B-aligned slices
  char* ws = (char*)d_ws;
  size_t off = 0;
  auto take = [&](size_t bytes) -> char* {
    char* p = ws + off;
    off += (bytes + 255) & ~(size_t)255;
    return p;
  };
  bf16* wqt = (bf16*)take((size_t)D_MODEL * D_MODEL * 2);  // [N][K] transposed
  bf16* wkt = (bf16*)take((size_t)D_MODEL * D_MODEL * 2);
  bf16* wvt = (bf16*)take((size_t)D_MODEL * D_MODEL * 2);
  bf16* wot = (bf16*)take((size_t)D_MODEL * D_MODEL * 2);
  bf16* w1t = (bf16*)take((size_t)D_MODEL * D_FF * 2);     // [4096][1024]
  bf16* w2t = (bf16*)take((size_t)D_FF * D_MODEL * 2);     // [1024][4096]
  bf16* n1  = (bf16*)take((size_t)MROWS * D_MODEL * 2);
  bf16* qb  = (bf16*)take((size_t)MROWS * D_MODEL * 2);
  bf16* kb  = (bf16*)take((size_t)MROWS * D_MODEL * 2);
  bf16* vb  = (bf16*)take((size_t)MROWS * D_MODEL * 2);
  bf16* vt  = (bf16*)take((size_t)MROWS * D_MODEL * 2);    // V^T per head
  bf16* ab  = (bf16*)take((size_t)MROWS * D_MODEL * 2);
  float* x1 = (float*)take((size_t)MROWS * D_MODEL * 4);
  bf16* n2  = (bf16*)take((size_t)MROWS * D_MODEL * 2);
  bf16* hb  = (bf16*)take((size_t)MROWS * D_FF * 2);

  // 1) weights -> bf16, transposed to [N][K]
  const dim3 tposeSq(D_MODEL / 64, D_MODEL / 64);
  cvt_transpose_bf16<<<tposeSq, dim3(256), 0, stream>>>(wq, wqt, D_MODEL, D_MODEL);
  cvt_transpose_bf16<<<tposeSq, dim3(256), 0, stream>>>(wk, wkt, D_MODEL, D_MODEL);
  cvt_transpose_bf16<<<tposeSq, dim3(256), 0, stream>>>(wv, wvt, D_MODEL, D_MODEL);
  cvt_transpose_bf16<<<tposeSq, dim3(256), 0, stream>>>(wo, wot, D_MODEL, D_MODEL);
  cvt_transpose_bf16<<<dim3(D_FF / 64, D_MODEL / 64), dim3(256), 0, stream>>>(
      w1, w1t, D_MODEL, D_FF);
  cvt_transpose_bf16<<<dim3(D_MODEL / 64, D_FF / 64), dim3(256), 0, stream>>>(
      w2, w2t, D_FF, D_MODEL);

  // 2) LN1
  ln_bf16<<<dim3(MROWS), dim3(256), 0, stream>>>(x, g1, be1, n1);

  // 3) Q,K,V projections
  const dim3 gProj(D_MODEL / 128, MROWS / 128);
  gemm_bf16<0><<<gProj, dim3(256), 0, stream>>>(n1, wqt, bq, nullptr, qb, nullptr,
                                                MROWS, D_MODEL, D_MODEL);
  gemm_bf16<0><<<gProj, dim3(256), 0, stream>>>(n1, wkt, bk, nullptr, kb, nullptr,
                                                MROWS, D_MODEL, D_MODEL);
  gemm_bf16<0><<<gProj, dim3(256), 0, stream>>>(n1, wvt, bv, nullptr, vb, nullptr,
                                                MROWS, D_MODEL, D_MODEL);

  // 3b) per-head V transpose for b128-friendly PV fragments
  transpose_v<<<dim3(SEQ / 64, BATCH * N_HEADS), dim3(128), 0, stream>>>(vb, vt);

  // 4) causal flash attention
  attn_kernel<<<dim3(SEQ / 64, BATCH * N_HEADS), dim3(128), 0, stream>>>(qb, kb, vt, ab);

  // 5) output projection + residual (f32)
  gemm_bf16<2><<<gProj, dim3(256), 0, stream>>>(ab, wot, bo, x, nullptr, x1,
                                                MROWS, D_MODEL, D_MODEL);

  // 6) LN2
  ln_bf16<<<dim3(MROWS), dim3(256), 0, stream>>>(x1, g2, be2, n2);

  // 7) MLP up + exact GELU
  gemm_bf16<1><<<dim3(D_FF / 128, MROWS / 128), dim3(256), 0, stream>>>(
      n2, w1t, b1, nullptr, hb, nullptr, MROWS, D_FF, D_MODEL);

  // 8) MLP down + residual -> final output
  gemm_bf16<2><<<gProj, dim3(256), 0, stream>>>(hb, w2t, b2, x1, nullptr, out,
                                                MROWS, D_MODEL, D_FF);
}
